// FeatureDetector_13709535608969
// MI455X (gfx1250) — matrix operand order repaired and checked by reading, not compiled, not run on hardware
//
#include <hip/hip_runtime.h>
#include <hip/hip_bf16.h>
#include <cmath>

#define PS 41
#define KPAD 1696                       /* 53 * 32, >= PS*PS = 1681 */
#define NFEAT 1000
#define NOCT 3
#define OPS 19
#define NBINS 36
#define BB 2
#define TWO_PI_F 6.28318530717958647692f
#define SIGMA0F 1.6f

#define OUT_LAFS 0
#define OUT_DESC (BB * NFEAT * 6)                    /* 12000  */
#define OUT_RESP (OUT_DESC + BB * NFEAT * 128)       /* 268000 */

typedef __attribute__((ext_vector_type(16))) _Float16 v16h;
typedef __attribute__((ext_vector_type(8)))  _Float16 v8h;
typedef __attribute__((ext_vector_type(8)))  float    v8f;

__device__ __forceinline__ int reflidx(int i, int n) {
    if (i < 0) i = -i;
    if (i >= n) i = 2 * n - 2 - i;
    return i;
}

__device__ __forceinline__ float bilin(const float* __restrict__ img, float xs, float ys,
                                       int H, int W) {
    float x0f = floorf(xs), y0f = floorf(ys);
    float fx = xs - x0f, fy = ys - y0f;
    int x0 = min(max((int)x0f, 0), W - 1);
    int x1 = min(x0 + 1, W - 1);
    int y0 = min(max((int)y0f, 0), H - 1);
    int y1 = min(y0 + 1, H - 1);
    float v00 = img[(size_t)y0 * W + x0];
    float v01 = img[(size_t)y0 * W + x1];
    float v10 = img[(size_t)y1 * W + x0];
    float v11 = img[(size_t)y1 * W + x1];
    return v00 * (1.f - fx) * (1.f - fy) + v01 * fx * (1.f - fy) +
           v10 * (1.f - fx) * fy + v11 * fx * fy;
}

// ---------------- Gaussian blur (separable, reflect pad) -------------------

__global__ void k_blur_h(const float* __restrict__ gray, float* __restrict__ tmp,
                         int H, int W, int oct, float sigma) {
    __shared__ float kc[32];
    __shared__ int rr;
    if (threadIdx.x == 0) {
        int r = (int)(3.0f * sigma + 0.5f);
        if (r < 1) r = 1;
        float sum = 0.f;
        for (int i = -r; i <= r; i++) {
            float q = (float)i / sigma;
            kc[i + r] = expf(-0.5f * q * q);
            sum += kc[i + r];
        }
        for (int i = 0; i <= 2 * r; i++) kc[i] /= sum;
        rr = r;
    }
    __syncthreads();
    int r = rr;
    const float* src = gray + (size_t)blockIdx.z * (1024u * 1024u);
    float* dst = tmp + (size_t)blockIdx.z * (1024u * 1024u);
    int id = blockIdx.x * blockDim.x + threadIdx.x;
    if (id >= H * W) return;
    int y = id / W, x = id - y * W;
    const float* row = src + ((size_t)(y << oct)) * 1024u;
    __builtin_prefetch(row + (size_t)(x << oct), 0, 1);
    float acc = 0.f;
    for (int d = -r; d <= r; d++) {
        int xx = reflidx(x + d, W);
        acc += kc[d + r] * row[(size_t)(xx << oct)];
    }
    dst[id] = acc;
}

__global__ void k_blur_v(const float* __restrict__ tmp, float* __restrict__ cur,
                         const float* __restrict__ prev, float* __restrict__ dog,
                         int H, int W, float sigma, int sidx) {
    __shared__ float kc[32];
    __shared__ int rr;
    if (threadIdx.x == 0) {
        int r = (int)(3.0f * sigma + 0.5f);
        if (r < 1) r = 1;
        float sum = 0.f;
        for (int i = -r; i <= r; i++) {
            float q = (float)i / sigma;
            kc[i + r] = expf(-0.5f * q * q);
            sum += kc[i + r];
        }
        for (int i = 0; i <= 2 * r; i++) kc[i] /= sum;
        rr = r;
    }
    __syncthreads();
    int r = rr;
    size_t bo = (size_t)blockIdx.z * (1024u * 1024u);
    const float* src = tmp + bo;
    float* dst = cur + bo;
    int id = blockIdx.x * blockDim.x + threadIdx.x;
    if (id >= H * W) return;
    int y = id / W, x = id - y * W;
    float acc = 0.f;
    for (int d = -r; d <= r; d++) {
        int yy = reflidx(y + d, H);
        acc += kc[d + r] * src[(size_t)yy * W + x];
    }
    dst[id] = acc;
    if (sidx >= 1) {
        float* dg = dog + (size_t)blockIdx.z * 5u * H * W + (size_t)(sidx - 1) * H * W;
        dg[id] = acc - prev[bo + id];
    }
}

// ------------- DoG extrema + per-chunk argmax (approx. per-octave top-k) ----

__global__ void k_cand(const float* __restrict__ dog, int H, int W, int oct,
                       float* __restrict__ candScore, int* __restrict__ candIdx) {
    int b = blockIdx.z;
    const float* dg = dog + (size_t)b * 5u * H * W;
    int HW = H * W;
    int N = 5 * HW;
    int chunk = (N + NFEAT - 1) / NFEAT;
    int start = blockIdx.x * chunk;
    int end = min(start + chunk, N);
    float best = -1.f;
    int bidx = start;
    for (int e = start + (int)threadIdx.x; e < end; e += blockDim.x) {
        int s = e / HW;
        int rem = e - s * HW;
        int y = rem / W;
        int x = rem - y * W;
        float v = dg[e];
        bool ismax = true, ismin = true;
        int s0 = max(s - 1, 0), s1 = min(s + 1, 4);
        int y0 = max(y - 1, 0), y1 = min(y + 1, H - 1);
        int x0 = max(x - 1, 0), x1 = min(x + 1, W - 1);
        for (int ds = s0; ds <= s1; ds++)
            for (int dy = y0; dy <= y1; dy++)
                for (int dx = x0; dx <= x1; dx++) {
                    float nv = dg[(size_t)(ds * H + dy) * W + dx];
                    ismax = ismax && (v >= nv);
                    ismin = ismin && (v <= nv);
                }
        float sc = (ismax || ismin) ? fabsf(v) : 0.f;
        if (sc > best || (sc == best && e < bidx)) { best = sc; bidx = e; }
    }
    __shared__ float ss[256];
    __shared__ int si[256];
    ss[threadIdx.x] = best;
    si[threadIdx.x] = bidx;
    __syncthreads();
    for (int off = 128; off > 0; off >>= 1) {
        if ((int)threadIdx.x < off) {
            float os = ss[threadIdx.x + off];
            int oi = si[threadIdx.x + off];
            if (os > ss[threadIdx.x] || (os == ss[threadIdx.x] && oi < si[threadIdx.x])) {
                ss[threadIdx.x] = os;
                si[threadIdx.x] = oi;
            }
        }
        __syncthreads();
    }
    if (threadIdx.x == 0) {
        int slot = b * 3 * NFEAT + oct * NFEAT + blockIdx.x;
        candScore[slot] = ss[0];
        candIdx[slot] = si[0];
    }
}

// ---------------- Subpixel refinement (3x3 Cramer solve) -------------------

__global__ void k_refine(const float* __restrict__ dog0, const float* __restrict__ dog1,
                         const float* __restrict__ dog2, const int* __restrict__ candIdx,
                         float* __restrict__ refX, float* __restrict__ refY,
                         float* __restrict__ refS, int totc) {
    int t = blockIdx.x * blockDim.x + threadIdx.x;
    if (t >= totc) return;
    int b = t / (3 * NFEAT);
    int r = t - b * 3 * NFEAT;
    int oct = r / NFEAT;
    int H = 1024 >> oct, W = H;
    const float* dg = (oct == 0 ? dog0 : (oct == 1 ? dog1 : dog2)) + (size_t)b * 5u * H * W;
    int HW = H * W;
    int e = candIdx[t];
    int s = e / HW;
    int rem = e - s * HW;
    int y = rem / W;
    int x = rem - y * W;
    s = min(max(s, 1), 3);
    y = min(max(y, 1), H - 2);
    x = min(max(x, 1), W - 2);
    float c[3][3][3];
    for (int i = 0; i < 3; i++)
        for (int j = 0; j < 3; j++)
            for (int k = 0; k < 3; k++)
                c[i][j][k] = dg[(size_t)((s + i - 1) * H + (y + j - 1)) * W + (x + k - 1)];
    float g0 = (c[2][1][1] - c[0][1][1]) * 0.5f;
    float g1 = (c[1][2][1] - c[1][0][1]) * 0.5f;
    float g2 = (c[1][1][2] - c[1][1][0]) * 0.5f;
    float dss = c[2][1][1] - 2.f * c[1][1][1] + c[0][1][1];
    float dyy = c[1][2][1] - 2.f * c[1][1][1] + c[1][0][1];
    float dxx = c[1][1][2] - 2.f * c[1][1][1] + c[1][1][0];
    float dsy = (c[2][2][1] - c[2][0][1] - c[0][2][1] + c[0][0][1]) * 0.25f;
    float dsx = (c[2][1][2] - c[2][1][0] - c[0][1][2] + c[0][1][0]) * 0.25f;
    float dyx = (c[1][2][2] - c[1][2][0] - c[1][0][2] + c[1][0][0]) * 0.25f;
    float a00 = dss + 1e-6f, a01 = dsy, a02 = dsx;
    float a10 = dsy, a11 = dyy + 1e-6f, a12 = dyx;
    float a20 = dsx, a21 = dyx, a22 = dxx + 1e-6f;
    float det = a00 * (a11 * a22 - a12 * a21) - a01 * (a10 * a22 - a12 * a20) +
                a02 * (a10 * a21 - a11 * a20);
    float o0 = 0.f, o1 = 0.f, o2 = 0.f;
    if (fabsf(det) > 1e-20f) {
        float inv = 1.f / det;
        float d0 = g0 * (a11 * a22 - a12 * a21) - a01 * (g1 * a22 - a12 * g2) +
                   a02 * (g1 * a21 - a11 * g2);
        float d1 = a00 * (g1 * a22 - a12 * g2) - g0 * (a10 * a22 - a12 * a20) +
                   a02 * (a10 * g2 - g1 * a20);
        float d2 = a00 * (a11 * g2 - g1 * a21) - a01 * (a10 * g2 - g1 * a20) +
                   g0 * (a10 * a21 - a11 * a20);
        o0 = -d0 * inv; o1 = -d1 * inv; o2 = -d2 * inv;
    }
    o0 = fminf(fmaxf(o0, -0.5f), 0.5f);
    o1 = fminf(fmaxf(o1, -0.5f), 0.5f);
    o2 = fminf(fmaxf(o2, -0.5f), 0.5f);
    float m = (float)(1 << oct);
    refX[t] = ((float)x + o2) * m;
    refY[t] = ((float)y + o1) * m;
    refS[t] = SIGMA0F * exp2f(((float)s + o0) / 3.0f) * m;
}

// ---------------- Global top-1000 via bitonic sort (per batch) -------------

#define SORTN 4096

__global__ void k_sort(const float* __restrict__ candScore, const float* __restrict__ refX,
                       const float* __restrict__ refY, const float* __restrict__ refS,
                       float* __restrict__ featX, float* __restrict__ featY,
                       float* __restrict__ featS, float* __restrict__ out) {
    __shared__ float key[SORTN];
    __shared__ int val[SORTN];
    int b = blockIdx.x;
    for (int i = threadIdx.x; i < SORTN; i += blockDim.x) {
        if (i < 3 * NFEAT) {
            key[i] = candScore[b * 3 * NFEAT + i];
        } else {
            key[i] = -3.4e38f;
        }
        val[i] = i;
    }
    __syncthreads();
    for (int k = 2; k <= SORTN; k <<= 1) {
        for (int j = k >> 1; j > 0; j >>= 1) {
            for (int t = threadIdx.x; t < SORTN; t += blockDim.x) {
                int ixj = t ^ j;
                if (ixj > t) {
                    bool dirDesc = ((t & k) == 0);
                    float ka = key[t], kb = key[ixj];
                    int va = val[t], vb = val[ixj];
                    bool aFirst = (ka > kb) || (ka == kb && va < vb);
                    bool doswap = dirDesc ? (!aFirst) : aFirst;
                    if (doswap) {
                        key[t] = kb; key[ixj] = ka;
                        val[t] = vb; val[ixj] = va;
                    }
                }
            }
            __syncthreads();
        }
    }
    for (int t = threadIdx.x; t < NFEAT; t += blockDim.x) {
        int c = val[t];
        featX[b * NFEAT + t] = refX[b * 3 * NFEAT + c];
        featY[b * NFEAT + t] = refY[b * 3 * NFEAT + c];
        featS[b * NFEAT + t] = refS[b * 3 * NFEAT + c];
        out[OUT_RESP + b * NFEAT + t] = key[t];
    }
}

// ---------------- Orientation histogram -> LAFs ----------------------------

__global__ void k_orient(const float* __restrict__ gray, const float* __restrict__ featX,
                         const float* __restrict__ featY, const float* __restrict__ featS,
                         float* __restrict__ out) {
    int f = blockIdx.x, b = blockIdx.y;
    int gi = b * NFEAT + f;
    float x = featX[gi], y = featY[gi], sig = featS[gi];
    const float* img = gray + (size_t)b * (1024u * 1024u);
    float rad = 6.0f * sig * 0.5f;
    __shared__ float op[OPS * OPS];
    __shared__ float hist[NBINS];
    __shared__ float hist2[NBINS];
    int lane = threadIdx.x;
    for (int t = lane; t < OPS * OPS; t += 32) {
        int i = t / OPS, j = t - i * OPS;
        float gu = -1.f + 2.f * (float)j / (float)(OPS - 1);
        float gv = -1.f + 2.f * (float)i / (float)(OPS - 1);
        op[t] = bilin(img, x + gu * rad, y + gv * rad, 1024, 1024);
    }
    for (int t = lane; t < NBINS; t += 32) hist[t] = 0.f;
    __syncthreads();
    for (int t = lane; t < OPS * OPS; t += 32) {
        int i = t / OPS, j = t - i * OPS;
        float gx = (op[i * OPS + min(j + 1, OPS - 1)] - op[i * OPS + max(j - 1, 0)]) * 0.5f;
        float gy = (op[min(i + 1, OPS - 1) * OPS + j] - op[max(i - 1, 0) * OPS + j]) * 0.5f;
        float mag = sqrtf(gx * gx + gy * gy + 1e-10f);
        float ang = atan2f(gy, gx);
        if (ang < 0.f) ang += TWO_PI_F;
        float gu = -1.f + 2.f * (float)j / (float)(OPS - 1);
        float gv = -1.f + 2.f * (float)i / (float)(OPS - 1);
        float w = mag * expf(-(gu * gu + gv * gv) * 2.0f);
        float a = ang * ((float)NBINS / TWO_PI_F);
        float a0 = floorf(a);
        float fr = a - a0;
        int i0 = ((int)a0) % NBINS;
        if (i0 < 0) i0 += NBINS;
        int i1 = (i0 + 1) % NBINS;
        atomicAdd(&hist[i0], w * (1.f - fr));
        atomicAdd(&hist[i1], w * fr);
    }
    __syncthreads();
    if (lane < NBINS) {
        const float ker[5] = {1.f / 16.f, 4.f / 16.f, 6.f / 16.f, 4.f / 16.f, 1.f / 16.f};
        float sm = 0.f;
        for (int d = -2; d <= 2; d++) {
            int idx = lane - d;
            idx = ((idx % NBINS) + NBINS) % NBINS;
            sm += ker[d + 2] * hist[idx];
        }
        hist2[lane] = sm;
    }
    __syncthreads();
    if (lane == 0) {
        int best = 0;
        float bv = hist2[0];
        for (int i = 1; i < NBINS; i++)
            if (hist2[i] > bv) { bv = hist2[i]; best = i; }
        float ang0 = (float)best * (TWO_PI_F / (float)NBINS);
        float ca = cosf(ang0), sa = sinf(ang0);
        float* L = out + OUT_LAFS + (size_t)gi * 6u;
        L[0] = rad * ca; L[1] = -rad * sa; L[2] = x;
        L[3] = rad * sa; L[4] = rad * ca;  L[5] = y;
    }
}

// ---------------- Descriptor: staged LDS tiles -> WMMA einsum --------------
// desc[s,o] = sum_p (wm[p]*Wsp[p,s]) * Wo[p,o]
// Phase 1: 41x41 bilinear patch -> LDS.
// Phase 2: materialize A (16 rows) and B (8 rows + zero row) in K-major f16
//          rows of length KPAD, matching the CDNA5 wave32 WMMA lane layout.
// Phase 3: 53 chunks: 2x ds_load_b128 (A) + 2x ds_load_b128 (B) + one
//          v_wmma_f32_16x16x32_f16, f32 accumulation.
//
// Dynamic LDS layout (bytes):
//   patch : [0, 6784)            1696 floats
//   Al    : [6784, 61056)        16 * KPAD halves
//   Bl    : [61056, 91584)       9  * KPAD halves (row 8 zeroed)
//   dsc   : [91584, 92096)       128 floats
#define SM_PATCH 0
#define SM_AL    6784
#define SM_BL    61056
#define SM_DSC   91584
#define SM_TOTAL 92096

__global__ void k_desc(const float* __restrict__ gray, float* __restrict__ out) {
    extern __shared__ __align__(64) char smem[];
    float*    patch = (float*)(smem + SM_PATCH);
    _Float16* Al    = (_Float16*)(smem + SM_AL);
    _Float16* Bl    = (_Float16*)(smem + SM_BL);
    float*    dsc   = (float*)(smem + SM_DSC);

    int f = blockIdx.x, b = blockIdx.y;
    int gi = b * NFEAT + f;
    const float* L = out + OUT_LAFS + (size_t)gi * 6u;
    float a00 = L[0], a01 = L[1], a02 = L[2];
    float a10 = L[3], a11 = L[4], a12 = L[5];
    const float* img = gray + (size_t)b * (1024u * 1024u);
    int lane = threadIdx.x;

    // Phase 1: sample patch
    for (int t = lane; t < PS * PS; t += 32) {
        int i = t / PS, j = t - i * PS;
        float pu = -1.f + 2.f * (float)j / (float)(PS - 1);
        float pv = -1.f + 2.f * (float)i / (float)(PS - 1);
        float px = a00 * pu + a01 * pv + a02;
        float py = a10 * pu + a11 * pv + a12;
        patch[t] = bilin(img, px, py, 1024, 1024);
    }
    // zero pad columns + B zero-row
    for (int t = lane; t < KPAD; t += 32) Bl[8 * KPAD + t] = (_Float16)0.f;
    for (int t = PS * PS + lane; t < KPAD; t += 32) {
        for (int s = 0; s < 16; s++) Al[s * KPAD + t] = (_Float16)0.f;
        for (int o = 0; o < 8; o++) Bl[o * KPAD + t] = (_Float16)0.f;
    }
    __syncthreads();

    // Phase 2: gradients -> A/B rows
    for (int t = lane; t < PS * PS; t += 32) {
        int i = t / PS, j = t - i * PS;
        float gx = (patch[i * PS + min(j + 1, PS - 1)] - patch[i * PS + max(j - 1, 0)]) * 0.5f;
        float gy = (patch[min(i + 1, PS - 1) * PS + j] - patch[max(i - 1, 0) * PS + j]) * 0.5f;
        float dmag = sqrtf(gx * gx + gy * gy + 1e-10f);
        float ang = atan2f(gy, gx);
        if (ang < 0.f) ang += TWO_PI_F;
        float cy = ((float)i - 20.f) / 20.5f;
        float cx = ((float)j - 20.f) / 20.5f;
        float wmv = dmag * expf(-(cy * cy + cx * cx) * 2.0f);
        float ob = ang * (8.f / TWO_PI_F);
        float o0 = floorf(ob);
        float fr = ob - o0;
        int j0 = ((int)o0) & 7;
        int j1 = (j0 + 1) & 7;

        float bcy = ((float)i + 0.5f) * (4.f / (float)PS) - 0.5f;
        float bcx = ((float)j + 0.5f) * (4.f / (float)PS) - 0.5f;
        float wyv[4], wxv[4];
#pragma unroll
        for (int q = 0; q < 4; q++) {
            wyv[q] = fmaxf(1.f - fabsf(bcy - (float)q), 0.f);
            wxv[q] = fmaxf(1.f - fabsf(bcx - (float)q), 0.f);
        }
#pragma unroll
        for (int s = 0; s < 16; s++)
            Al[s * KPAD + t] = (_Float16)(wmv * wyv[s >> 2] * wxv[s & 3]);
#pragma unroll
        for (int o = 0; o < 8; o++) {
            float bv = (o == j0) ? (1.f - fr) : ((o == j1) ? fr : 0.f);
            Bl[o * KPAD + t] = (_Float16)bv;
        }
    }
    __syncthreads();

    // Phase 3: WMMA K-loop.
    // A (16x32 f16): M = lane&15; VGPR v: K = (v>=4?16:0)+(lane>=16?8:0)+2*(v&3)+h
    // B (32x16 f16): N = lane&15; VGPR v: K = (v>=4?16:0)+(lane>=16?8:0)+2*(v&3)+h
    // D (16x16 f32): N = lane&15; VGPR i: M = i+(lane>=16?8:0)
    int srow = lane & 15;
    int ocol = lane & 15;
    int hiA = (lane >= 16) ? 8 : 0;
    int hiB = (lane >= 16) ? 16 : 0;
    const _Float16* Arow = Al + srow * KPAD;
    const _Float16* Brow = Bl + min(ocol, 8) * KPAD;

    v8f acc = {};
    for (int chunk = 0; chunk < KPAD / 32; chunk++) {
        int kb = chunk * 32;
        v8h alo = *(const v8h*)(Arow + kb + hiA);
        v8h ahi = *(const v8h*)(Arow + kb + 16 + hiA);
        v16h a = __builtin_shufflevector(alo, ahi, 0, 1, 2, 3, 4, 5, 6, 7,
                                         8, 9, 10, 11, 12, 13, 14, 15);
        v8h blo = *(const v8h*)(Brow + kb + hiA);
        v8h bhi = *(const v8h*)(Brow + kb + 16 + hiA);
        v16h bm = __builtin_shufflevector(blo, bhi, 0, 1, 2, 3, 4, 5, 6, 7,
                                          8, 9, 10, 11, 12, 13, 14, 15);
        acc = __builtin_amdgcn_wmma_f32_16x16x32_f16(
            false, a, false, bm, (short)0, acc, false, false);
    }

    if (ocol < 8) {
#pragma unroll
        for (int i = 0; i < 8; i++) {
            int s = i + hiA;
            dsc[s * 8 + ocol] = acc[i];
        }
    }
    __syncthreads();

    if (lane == 0) {
        float n = 0.f;
        for (int i = 0; i < 128; i++) n += dsc[i] * dsc[i];
        n = sqrtf(n) + 1e-8f;
        float n2 = 0.f;
        for (int i = 0; i < 128; i++) {
            float d = fminf(dsc[i] / n, 0.2f);
            dsc[i] = d;
            n2 += d * d;
        }
        n2 = sqrtf(n2) + 1e-8f;
        float ssum = 0.f;
        for (int i = 0; i < 128; i++) {
            dsc[i] /= n2;
            ssum += dsc[i];
        }
        ssum += 1e-8f;
        float* od = out + OUT_DESC + (size_t)gi * 128u;
        for (int i = 0; i < 128; i++) od[i] = sqrtf(dsc[i] / ssum + 1e-10f);
    }
}

// ---------------------------------------------------------------------------

extern "C" void kernel_launch(void* const* d_in, const int* in_sizes, int n_in,
                              void* d_out, int out_size, void* d_ws, size_t ws_size,
                              hipStream_t stream) {
    const float* gray = (const float*)d_in[0];
    float* out = (float*)d_out;
    float* ws = (float*)d_ws;
    const int B = BB;
    const size_t S0 = 1024u * 1024u, S1 = 512u * 512u, S2 = 256u * 256u;

    size_t off = 0;
    float* dog0 = ws + off; off += (size_t)B * 5u * S0;
    float* dog1 = ws + off; off += (size_t)B * 5u * S1;
    float* dog2 = ws + off; off += (size_t)B * 5u * S2;
    float* tmp  = ws + off; off += (size_t)B * S0;
    float* bufA = ws + off; off += (size_t)B * S0;
    float* bufB = ws + off; off += (size_t)B * S0;
    float* candScore = ws + off; off += (size_t)B * 3u * NFEAT;
    int*   candIdx = (int*)(ws + off); off += (size_t)B * 3u * NFEAT;
    float* refX = ws + off; off += (size_t)B * 3u * NFEAT;
    float* refY = ws + off; off += (size_t)B * 3u * NFEAT;
    float* refS = ws + off; off += (size_t)B * 3u * NFEAT;
    float* featX = ws + off; off += (size_t)B * NFEAT;
    float* featY = ws + off; off += (size_t)B * NFEAT;
    float* featS = ws + off; off += (size_t)B * NFEAT;
    (void)off; (void)ws_size; (void)in_sizes; (void)n_in; (void)out_size;

    float* dogs[3] = {dog0, dog1, dog2};
    for (int o = 0; o < NOCT; o++) {
        int H = 1024 >> o, W = H;
        float* prev = bufA;
        float* cur = bufB;
        for (int s = 0; s < 6; s++) {
            float sigma = SIGMA0F * exp2f((float)s / 3.0f);
            dim3 g((unsigned)((H * W + 255) / 256), 1, (unsigned)B);
            k_blur_h<<<g, 256, 0, stream>>>(gray, tmp, H, W, o, sigma);
            k_blur_v<<<g, 256, 0, stream>>>(tmp, cur, prev, dogs[o], H, W, sigma, s);
            float* t2 = prev; prev = cur; cur = t2;
        }
        dim3 gc(NFEAT, 1, (unsigned)B);
        k_cand<<<gc, 256, 0, stream>>>(dogs[o], H, W, o, candScore, candIdx);
    }

    int totc = B * 3 * NFEAT;
    k_refine<<<(totc + 255) / 256, 256, 0, stream>>>(dog0, dog1, dog2, candIdx,
                                                     refX, refY, refS, totc);
    k_sort<<<B, 1024, 0, stream>>>(candScore, refX, refY, refS,
                                   featX, featY, featS, out);

    dim3 gf(NFEAT, (unsigned)B);
    k_orient<<<gf, 32, 0, stream>>>(gray, featX, featY, featS, out);
    k_desc<<<gf, 32, SM_TOTAL, stream>>>(gray, out);
}
